// LMUCellGating_70927089926378
// MI455X (gfx1250) — matrix operand order repaired and checked by reading, not compiled, not run on hardware
//
#include <hip/hip_runtime.h>
#include <hip/hip_bf16.h>

// LMU cell for gfx1250 (MI455X):
//  - hoisted input GEMMs over all B*T rows (WMMA bf16, 16x64 strip per wave)
//  - persistent recurrent kernel, device-scope grid barrier, per-step WMMA
//    with the A row-block staged into LDS via async global->LDS loads.

#define BB 64
#define TT 512
#define DD 512
#define UU 1024
#define OO 256
#define NWG 64

typedef __attribute__((ext_vector_type(16))) __bf16 v16bf;
typedef __attribute__((ext_vector_type(8)))  __bf16 v8bf;
typedef __attribute__((ext_vector_type(8)))  float  v8f;
typedef __attribute__((ext_vector_type(4)))  int    vi4;

#if defined(__gfx1250__) && \
    __has_builtin(__builtin_amdgcn_global_load_async_to_lds_b128) && \
    __has_builtin(__builtin_amdgcn_s_wait_asynccnt)
#define USE_ASYNC_LDS 1
#else
#define USE_ASYNC_LDS 0
#endif

__device__ __forceinline__ unsigned short f32_bf16(float f) {
  union { float f; unsigned u; } v; v.f = f;
  unsigned u = v.u + 0x7FFFu + ((v.u >> 16) & 1u);   // round-to-nearest-even
  return (unsigned short)(u >> 16);
}
__device__ __forceinline__ float bf16_f32(unsigned short h) {
  union { unsigned u; float f; } v; v.u = ((unsigned)h) << 16; return v.f;
}

// A-fragment (16-bit A 16x32): lanes 0-15 row=lane,
// halves 0-15=K0..15; lanes 16-31 K16..31 (contiguous, matches B).
__device__ __forceinline__ v16bf a_frag(const unsigned short* pa, int k0) {
  return *(const v16bf*)(pa + k0);
}

// ---- recurrent-kernel tile: A staged in LDS (16 rows x K), B N-major ----
__device__ __forceinline__ v8f wmma_tile_s(const unsigned short* sA, int K,
                                           const unsigned short* Bt,
                                           int n0, int lane) {
  v8f c = {};
  const unsigned short* pa = sA + (size_t)(lane & 15) * K + ((lane < 16) ? 0 : 16);
  const unsigned short* pb = Bt + (size_t)(n0 + (lane & 15)) * K + ((lane < 16) ? 0 : 16);
  for (int k0 = 0; k0 < K; k0 += 32) {
    v16bf a = a_frag(pa, k0);
    v16bf b = *(const v16bf*)(pb + k0);
    c = __builtin_amdgcn_wmma_f32_16x16x32_bf16(false, a, false, b,
                                                (short)0, c, false, false);
  }
  return c;
}

// Stage 16 rows x K bf16 (row-major, contiguous) from global into LDS.
// Uses CDNA5 async global->LDS b128 copies when available.
__device__ __forceinline__ void stage16(const unsigned short* g,
                                        unsigned short* s, int K) {
  int chunks = 2 * K;                       // 16-byte chunks in 16*K*2 bytes
  for (int c = threadIdx.x; c < chunks; c += 256) {
    int e = c * 8;                          // element offset
#if USE_ASYNC_LDS
    __builtin_amdgcn_global_load_async_to_lds_b128((vi4*)(g + e),
                                                   (vi4*)(s + e), 0, 0);
#else
    *(int4*)(s + e) = *(const int4*)(g + e);
#endif
  }
#if USE_ASYNC_LDS
  __builtin_amdgcn_s_wait_asynccnt(0);
#endif
  __syncthreads();
}

// ---------------- conversion kernels ----------------
__global__ void __launch_bounds__(256) cvt_copy(const float* src,
                                                unsigned short* dst, int n) {
  int i = blockIdx.x * 256 + threadIdx.x;
  if (i < n) dst[i] = f32_bf16(src[i]);
}
// src K x N fp32 row-major -> dst N x K bf16 (transposed, N-major)
__global__ void __launch_bounds__(256) cvt_tr(const float* src,
                                              unsigned short* dst, int K, int N) {
  int i = blockIdx.x * 256 + threadIdx.x;
  if (i < K * N) {
    int k = i / N, n = i % N;
    dst[(size_t)n * K + k] = f32_bf16(src[i]);
  }
}

// ---------------- hoisted input GEMMs: 16x64 strip per wave ----------------
__global__ void __launch_bounds__(256) gemm_rows4(const unsigned short* Abf,
                                                  const unsigned short* Bt,
                                                  int K, int N, const float* bias,
                                                  float* out, int ngrp) {
  int wv = (blockIdx.x * 256 + threadIdx.x) >> 5;
  int lane = threadIdx.x & 31;
  int mt = wv / ngrp, ng = wv % ngrp;
  int n0 = ng * 64;
  const unsigned short* pa =
      Abf + (size_t)(mt * 16 + (lane & 15)) * K + ((lane < 16) ? 0 : 16);
  const unsigned short* pb =
      Bt + (size_t)(n0 + (lane & 15)) * K + ((lane < 16) ? 0 : 16);
  v8f c0 = {}, c1 = {}, c2 = {}, c3 = {};
  for (int k0 = 0; k0 < K; k0 += 32) {
    v16bf a = a_frag(pa, k0);            // shared by 4 N-tiles
    v16bf b0 = *(const v16bf*)(pb + k0);
    v16bf b1 = *(const v16bf*)(pb + (size_t)16 * K + k0);
    v16bf b2 = *(const v16bf*)(pb + (size_t)32 * K + k0);
    v16bf b3 = *(const v16bf*)(pb + (size_t)48 * K + k0);
    c0 = __builtin_amdgcn_wmma_f32_16x16x32_bf16(false, a, false, b0, (short)0, c0, false, false);
    c1 = __builtin_amdgcn_wmma_f32_16x16x32_bf16(false, a, false, b1, (short)0, c1, false, false);
    c2 = __builtin_amdgcn_wmma_f32_16x16x32_bf16(false, a, false, b2, (short)0, c2, false, false);
    c3 = __builtin_amdgcn_wmma_f32_16x16x32_bf16(false, a, false, b3, (short)0, c3, false, false);
  }
  int rb = mt * 16 + ((lane < 16) ? 0 : 8);
  int col = n0 + (lane & 15);
  v8f acc[4] = {c0, c1, c2, c3};
#pragma unroll
  for (int tl = 0; tl < 4; ++tl) {
    int n = col + tl * 16;
    float bi = bias ? bias[n] : 0.0f;
#pragma unroll
    for (int i = 0; i < 8; ++i) out[(size_t)(rb + i) * N + n] = acc[tl][i] + bi;
  }
}

// Xu[r] = dot(x[r,:], input_encoders)  (N==1; one wave per row)
__global__ void __launch_bounds__(256) dot_xu(const float* x, const float* ie,
                                              float* xu) {
  int wv = (blockIdx.x * 256 + threadIdx.x) >> 5;
  int lane = threadIdx.x & 31;
  const float* row = x + (size_t)wv * DD;
  float s = 0.f;
  for (int k = lane; k < DD; k += 32) s += row[k] * ie[k];
#pragma unroll
  for (int off = 16; off; off >>= 1) s += __shfl_xor(s, off, 32);
  if (lane == 0) xu[wv] = s;
}

__global__ void __launch_bounds__(256) init_state(float* mstate,
                                                  unsigned short* mb16,
                                                  unsigned short* hb16,
                                                  unsigned* sync) {
  int i = blockIdx.x * 256 + threadIdx.x;
  if (i < BB * OO) { mstate[i] = 0.f; mb16[i] = 0; }
  if (i < BB * UU) hb16[i] = 0;
  if (i < 2) sync[i] = 0;
}

// ---------------- persistent recurrent kernel ----------------
__device__ __forceinline__ void grid_sync(unsigned* sync, unsigned nb) {
  __syncthreads();
  if (threadIdx.x == 0) {
    unsigned* counter = sync;
    unsigned* gen = sync + 1;
    unsigned g = __hip_atomic_load(gen, __ATOMIC_RELAXED, __HIP_MEMORY_SCOPE_AGENT);
    unsigned arrived = __hip_atomic_fetch_add(counter, 1u, __ATOMIC_ACQ_REL,
                                              __HIP_MEMORY_SCOPE_AGENT);
    if (arrived == nb - 1u) {
      __hip_atomic_store(counter, 0u, __ATOMIC_RELAXED, __HIP_MEMORY_SCOPE_AGENT);
      __hip_atomic_fetch_add(gen, 1u, __ATOMIC_ACQ_REL, __HIP_MEMORY_SCOPE_AGENT);
    } else {
      while (__hip_atomic_load(gen, __ATOMIC_ACQUIRE,
                               __HIP_MEMORY_SCOPE_AGENT) == g)
        __builtin_amdgcn_s_sleep(2);
    }
  }
  __syncthreads();
}

struct RecurArgs {
  const unsigned short* WhhT;  // 1024x1024 bf16 (hidden_kernel^T)
  const unsigned short* WfhT;  // 256x1024  bf16 (forget_hidden^T)
  const unsigned short* WmT;   // 1024x256  bf16 (memory_kernel^T)
  const unsigned short* ATt;   // 256x256   bf16 (AT^T)
  const float* Xf;             // (B*T) x 256
  const float* Xu;             // (B*T)
  const float* he;             // hidden_encoders (UNITS)
  const float* me;             // memory_encoders (ORDER)
  const float* BT;             // (ORDER)
  float* hW;                   // 64x1024 scratch: h @ W_hh
  float* fbuf;                 // 64x256  scratch: f_t
  float* mAT;                  // 64x256  scratch: m @ AT
  float* ubuf;                 // 64
  float* mstate;               // 64x256 fp32 m
  unsigned short* mb16;        // 64x256 bf16 m
  unsigned short* hb16;        // 64x1024 bf16 h
  float* dout;                 // B x T x UNITS (holds x@W_ih, becomes hs)
  unsigned* sync;
};

__global__ void __launch_bounds__(256) lmu_recur(RecurArgs A) {
  __shared__ unsigned short sA[16 * UU];   // staged A row-block (32 KB)
  const int lane = threadIdx.x & 31;
  const int wv   = (blockIdx.x << 3) | (threadIdx.x >> 5);  // 0..511
  const int tid  = blockIdx.x * 256 + threadIdx.x;          // 0..16383
  const int blk  = blockIdx.x;

  for (int t = 0; t < TT; ++t) {
    // ---- phase 1: hW = h@Whh, f = Xf_t + h@Wfh, mAT = m@AT, u ----
    if (blk < 32) {                       // h @ W_hh -> hW (blocks share mt)
      int mt = blk >> 3;
      stage16(A.hb16 + (size_t)mt * 16 * UU, sA, UU);
      int nt = wv & 63;
      v8f c = wmma_tile_s(sA, UU, A.WhhT, nt * 16, lane);
      int rb = mt * 16 + ((lane < 16) ? 0 : 8);
      int n  = nt * 16 + (lane & 15);
#pragma unroll
      for (int i = 0; i < 8; ++i) A.hW[(rb + i) * UU + n] = c[i];
    } else if (blk < 40) {                // f = Xf_t + h @ W_fh
      int mt = (blk - 32) >> 1;
      stage16(A.hb16 + (size_t)mt * 16 * UU, sA, UU);
      int nt = (wv - 256) & 15;
      v8f c = wmma_tile_s(sA, UU, A.WfhT, nt * 16, lane);
      int rb = mt * 16 + ((lane < 16) ? 0 : 8);
      int o  = nt * 16 + (lane & 15);
#pragma unroll
      for (int i = 0; i < 8; ++i) {
        int b = rb + i;
        A.fbuf[b * OO + o] = c[i] + A.Xf[((size_t)b * TT + t) * OO + o];
      }
    } else if (blk < 48) {                // m @ AT
      int mt = (blk - 40) >> 1;
      stage16(A.mb16 + (size_t)mt * 16 * OO, sA, OO);
      int nt = (wv - 320) & 15;
      v8f c = wmma_tile_s(sA, OO, A.ATt, nt * 16, lane);
      int rb = mt * 16 + ((lane < 16) ? 0 : 8);
      int o  = nt * 16 + (lane & 15);
#pragma unroll
      for (int i = 0; i < 8; ++i) A.mAT[(rb + i) * OO + o] = c[i];
    } else if (blk < 56) {                // u = Xu_t + h.he + m.me
      int b = wv - 384;
      float s = A.Xu[(size_t)b * TT + t];
      for (int n = lane; n < UU; n += 32)
        s += bf16_f32(A.hb16[b * UU + n]) * A.he[n];
      for (int o = lane; o < OO; o += 32)
        s += A.mstate[b * OO + o] * A.me[o];
#pragma unroll
      for (int off = 16; off; off >>= 1) s += __shfl_xor(s, off, 32);
      if (lane == 0) A.ubuf[b] = s;
    }
    grid_sync(A.sync, NWG);

    // ---- phase 2: m = m + m@AT + f * (u * BT) ----
    {
      int b = tid >> 8, o = tid & 255;
      float mn = A.mstate[tid] + A.mAT[tid] + A.fbuf[tid] * A.ubuf[b] * A.BT[o];
      A.mstate[tid] = mn;
      A.mb16[tid] = f32_bf16(mn);
    }
    grid_sync(A.sync, NWG);

    // ---- phase 3: h = tanh(Xh_t + hW + m@Wm); write hs ----
    if (blk < 32) {
      int mt = blk >> 3;
      stage16(A.mb16 + (size_t)mt * 16 * OO, sA, OO);
      int nt = wv & 63;
      v8f c = wmma_tile_s(sA, OO, A.WmT, nt * 16, lane);
      int rb = mt * 16 + ((lane < 16) ? 0 : 8);
      int n  = nt * 16 + (lane & 15);
#pragma unroll
      for (int i = 0; i < 8; ++i) {
        int b = rb + i;
        size_t oi = ((size_t)b * TT + t) * (size_t)UU + n;
        float v = tanhf(A.dout[oi] + A.hW[b * UU + n] + c[i]);
        A.dout[oi] = v;
        A.hb16[b * UU + n] = f32_bf16(v);
      }
    }
    grid_sync(A.sync, NWG);
  }
}

extern "C" void kernel_launch(void* const* d_in, const int* in_sizes, int n_in,
                              void* d_out, int out_size, void* d_ws, size_t ws_size,
                              hipStream_t stream) {
  (void)in_sizes; (void)n_in; (void)out_size; (void)ws_size;
  const float* x   = (const float*)d_in[0];
  const float* ie  = (const float*)d_in[1];
  const float* he  = (const float*)d_in[2];
  const float* me  = (const float*)d_in[3];
  const float* Wih = (const float*)d_in[4];
  const float* Whh = (const float*)d_in[5];
  const float* Wmh = (const float*)d_in[6];
  const float* Wfi = (const float*)d_in[7];
  const float* Wfh = (const float*)d_in[8];
  const float* fb  = (const float*)d_in[9];
  const float* ATm = (const float*)d_in[10];
  const float* BTm = (const float*)d_in[11];
  float* dout = (float*)d_out;

  // -------- workspace carve-up --------
  char* w = (char*)d_ws;
  size_t off = 0;
  auto alloc = [&](size_t bytes) {
    size_t o = off; off += (bytes + 255) & ~(size_t)255; return o;
  };
  unsigned short* xb    = (unsigned short*)(w + alloc((size_t)BB * TT * DD * 2));
  unsigned short* WhT   = (unsigned short*)(w + alloc((size_t)UU * DD * 2));
  unsigned short* WfinT = (unsigned short*)(w + alloc((size_t)OO * DD * 2));
  unsigned short* WhhT  = (unsigned short*)(w + alloc((size_t)UU * UU * 2));
  unsigned short* WfhT  = (unsigned short*)(w + alloc((size_t)OO * UU * 2));
  unsigned short* WmT   = (unsigned short*)(w + alloc((size_t)UU * OO * 2));
  unsigned short* ATt   = (unsigned short*)(w + alloc((size_t)OO * OO * 2));
  float* Xf     = (float*)(w + alloc((size_t)BB * TT * OO * 4));
  float* Xu     = (float*)(w + alloc((size_t)BB * TT * 4));
  float* hW     = (float*)(w + alloc((size_t)BB * UU * 4));
  float* fbuf   = (float*)(w + alloc((size_t)BB * OO * 4));
  float* mAT    = (float*)(w + alloc((size_t)BB * OO * 4));
  float* ubuf   = (float*)(w + alloc((size_t)BB * 4));
  float* mstate = (float*)(w + alloc((size_t)BB * OO * 4));
  unsigned short* mb16 = (unsigned short*)(w + alloc((size_t)BB * OO * 2));
  unsigned short* hb16 = (unsigned short*)(w + alloc((size_t)BB * UU * 2));
  unsigned* sync = (unsigned*)(w + alloc(2 * sizeof(unsigned)));

  // -------- convert inputs/weights to bf16 (weights transposed) --------
  {
    int n = BB * TT * DD;
    cvt_copy<<<n / 256, 256, 0, stream>>>(x, xb, n);
  }
  cvt_tr<<<(DD * UU) / 256, 256, 0, stream>>>(Wih, WhT, DD, UU);
  cvt_tr<<<(UU * UU) / 256, 256, 0, stream>>>(Whh, WhhT, UU, UU);
  cvt_tr<<<(OO * UU) / 256, 256, 0, stream>>>(Wmh, WmT, OO, UU);
  cvt_tr<<<(DD * OO) / 256, 256, 0, stream>>>(Wfi, WfinT, DD, OO);
  cvt_tr<<<(UU * OO) / 256, 256, 0, stream>>>(Wfh, WfhT, UU, OO);
  cvt_tr<<<(OO * OO) / 256, 256, 0, stream>>>(ATm, ATt, OO, OO);

  // -------- hoisted x-projections (parallel over all B*T rows) --------
  // d_out <- x @ input_kernel   (M=32768, N=1024, K=512); 16x64 per wave
  {
    int waves = (BB * TT / 16) * (UU / 64);
    gemm_rows4<<<waves * 32 / 256, 256, 0, stream>>>(xb, WhT, DD, UU, nullptr,
                                                     dout, UU / 64);
  }
  // Xf <- x @ forget_input_kernel + forget_bias  (N=256)
  {
    int waves = (BB * TT / 16) * (OO / 64);
    gemm_rows4<<<waves * 32 / 256, 256, 0, stream>>>(xb, WfinT, DD, OO, fb,
                                                     Xf, OO / 64);
  }
  // Xu <- x @ input_encoders
  dot_xu<<<(BB * TT) * 32 / 256, 256, 0, stream>>>(x, ie, Xu);

  // -------- init recurrent state + barrier vars --------
  init_state<<<(BB * UU) / 256, 256, 0, stream>>>(mstate, mb16, hb16, sync);

  // -------- persistent recurrence over T=512 --------
  RecurArgs A;
  A.WhhT = WhhT; A.WfhT = WfhT; A.WmT = WmT; A.ATt = ATt;
  A.Xf = Xf; A.Xu = Xu; A.he = he; A.me = me; A.BT = BTm;
  A.hW = hW; A.fbuf = fbuf; A.mAT = mAT; A.ubuf = ubuf;
  A.mstate = mstate; A.mb16 = mb16; A.hb16 = hb16;
  A.dout = dout; A.sync = sync;
  lmu_recur<<<NWG, 256, 0, stream>>>(A);
}
